// DecorrLinear_781684048736
// MI455X (gfx1250) — matrix operand order repaired and checked by reading, not compiled, not run on hardware
//
#include <hip/hip_runtime.h>

// ---------------- problem dims (fixed by the reference) ----------------
#define BB     8
#define LL     4096
#define DD     1024
#define OO     2048
#define NSAMP  1024
#define NR     (BB * NSAMP)   // 8192 sampled rows
#define MROWS  (BB * LL)      // 32768 rows of y
#define KAPPA  0.5f

typedef __attribute__((ext_vector_type(2)))  float  v2f;
typedef __attribute__((ext_vector_type(4)))  float  v4f;
typedef __attribute__((ext_vector_type(8)))  float  v8f;
typedef __attribute__((ext_vector_type(8)))  __bf16 v8bf;
typedef __attribute__((ext_vector_type(16))) __bf16 v16bf;

typedef unsigned int u32x4 __attribute__((ext_vector_type(4)));
typedef int          i32x4 __attribute__((ext_vector_type(4)));
typedef int          i32x8 __attribute__((ext_vector_type(8)));

#if __has_builtin(__builtin_amdgcn_tensor_load_to_lds) && \
    __has_builtin(__builtin_amdgcn_s_wait_tensorcnt)
#define USE_TDM 1
#else
#define USE_TDM 0
#endif

static __device__ __forceinline__ __bf16 f2bf(float f) {
  unsigned u = __builtin_bit_cast(unsigned, f);
  unsigned r = u + 0x7FFFu + ((u >> 16) & 1u);   // round-to-nearest-even
  unsigned short h = (unsigned short)(r >> 16);
  return __builtin_bit_cast(__bf16, h);
}

// A fragment (16x32 bf16, ISA 7.12.2): lane m=lane&15; half=lane>>4 holds
// K = [8*half, +8) in v[0..7] and K = [16+8*half, +8) in v[8..15].
static __device__ __forceinline__ v16bf load_a_bf16(const __bf16* __restrict__ row, int lm) {
  v8bf lo = *(const v8bf*)(row + lm * 8);
  v8bf hi = *(const v8bf*)(row + 16 + lm * 8);
  return __builtin_shufflevector(lo, hi, 0, 1, 2, 3, 4, 5, 6, 7,
                                 8, 9, 10, 11, 12, 13, 14, 15);
}

// B fragment (32x16 bf16): lane n=lane&15; half=lane>>4 holds K = [8*half, +8)
// in v[0..7] and K = [16+8*half, +8) in v[8..15]; column n of B == row n of row-major fused.
static __device__ __forceinline__ v16bf load_b_bf16(const __bf16* __restrict__ row, int lm) {
  const __bf16* q = row + lm * 8;
  v8bf lo = *(const v8bf*)(q);
  v8bf hi = *(const v8bf*)(q + 16);
  return __builtin_shufflevector(lo, hi, 0, 1, 2, 3, 4, 5, 6, 7,
                                 8, 9, 10, 11, 12, 13, 14, 15);
}

#if USE_TDM
// Issue one TDM 2D tile load: rows x 32 bf16 elements, row stride DD elements,
// packed densely into LDS at lds_off.  D# packing per CDNA5 ISA.
// This toolchain's builtin takes 6 args (g0,g1,g2,g3,g4,cpol).
static __device__ __forceinline__ void tdm_load_tile(const __bf16* gsrc,
                                                     unsigned lds_off, int rows) {
  unsigned long long ga = (unsigned long long)(size_t)gsrc;
  u32x4 g0;
  g0[0] = 1u;                                   // count=1, user mode, no gather
  g0[1] = lds_off;                              // lds_addr (bytes)
  g0[2] = (unsigned)ga;                         // global_addr[31:0]
  g0[3] = (unsigned)(ga >> 32) | (2u << 30);    // global_addr[56:32] | type=2
  i32x8 g1;
  g1[0] = 1 << 16;                              // workgroup_mask=0, data_size=1 (2B)
  g1[1] = (int)((unsigned)DD << 16);            // tensor_dim0[15:0] in bits[63:48]
  g1[2] = 0;                                    // td0 hi=0, td1[15:0]=0 (td1=1<<20)
  g1[3] = (int)(16u | (32u << 16));             // td1[31:16]=16, tile_dim0=32
  g1[4] = rows;                                 // tile_dim1=rows, tile_dim2=0
  g1[5] = DD;                                   // tensor_dim0_stride[31:0]
  g1[6] = 0;                                    // stride hi, dim1_stride lo
  g1[7] = 0;
  i32x4 gz  = {0, 0, 0, 0};                     // 2D: groups 2/3 unused
  i32x8 gz8 = {0, 0, 0, 0, 0, 0, 0, 0};
  __builtin_amdgcn_tensor_load_to_lds(g0, g1, gz, gz, gz8, 0);
}
#endif

// ---------------- x (f32) -> x (bf16) streaming convert ----------------
__global__ void convert_x_kernel(const float* __restrict__ x, __bf16* __restrict__ xb) {
  size_t i = ((size_t)blockIdx.x * blockDim.x + threadIdx.x) * 8;
  v4f a = *(const v4f*)(x + i);
  v4f b = *(const v4f*)(x + i + 4);
  v8bf o;
  o[0] = f2bf(a[0]); o[1] = f2bf(a[1]); o[2] = f2bf(a[2]); o[3] = f2bf(a[3]);
  o[4] = f2bf(b[0]); o[5] = f2bf(b[1]); o[6] = f2bf(b[2]); o[7] = f2bf(b[3]);
  *(v8bf*)(xb + i) = o;
}

// ---------------- fused = W @ Q (f32 WMMA), emitted as bf16 ----------------
__global__ void fused_gemm_kernel(const float* __restrict__ W,
                                  const float* __restrict__ Q,
                                  __bf16* __restrict__ fusedB) {
  const int lane = threadIdx.x & 31;
  const int wv   = threadIdx.x >> 5;
  const int lm   = lane >> 4, ln = lane & 15;
  const int row0 = blockIdx.x * 128 + (wv >> 1) * 32;
  const int col0 = blockIdx.y * 64  + (wv & 1) * 32;

  const float* wr0 = W + (size_t)(row0 + ln) * DD;
  const float* wr1 = wr0 + (size_t)16 * DD;

  v8f acc00 = {}, acc01 = {}, acc10 = {}, acc11 = {};
#pragma unroll 2
  for (int k = 0; k < DD; k += 4) {
    const int ka = k + lm * 2;
    v2f a0 = *(const v2f*)(wr0 + ka);
    v2f a1 = *(const v2f*)(wr1 + ka);
    const float* q0 = Q + (size_t)ka * DD + (col0 + ln);
    v2f b0 = { q0[0],  q0[DD] };
    v2f b1 = { q0[16], q0[DD + 16] };
    acc00 = __builtin_amdgcn_wmma_f32_16x16x4_f32(false, a0, false, b0, (short)0, acc00, false, false);
    acc01 = __builtin_amdgcn_wmma_f32_16x16x4_f32(false, a0, false, b1, (short)0, acc01, false, false);
    acc10 = __builtin_amdgcn_wmma_f32_16x16x4_f32(false, a1, false, b0, (short)0, acc10, false, false);
    acc11 = __builtin_amdgcn_wmma_f32_16x16x4_f32(false, a1, false, b1, (short)0, acc11, false, false);
  }

  v8f acc[2][2] = { { acc00, acc01 }, { acc10, acc11 } };
#pragma unroll
  for (int i = 0; i < 2; ++i)
#pragma unroll
    for (int j = 0; j < 2; ++j) {
      __bf16* out = fusedB + (size_t)(row0 + i * 16 + lm * 8) * DD + (col0 + j * 16 + ln);
#pragma unroll
      for (int r = 0; r < 8; ++r) out[(size_t)r * DD] = f2bf(acc[i][j][r]);
    }
}

// ---- y = xB @ fusedBᵀ + bias: TDM-staged, LDS double-buffered bf16 WMMA ----
__global__ void __launch_bounds__(256) y_gemm_kernel(const __bf16* __restrict__ xb,
                                                     const __bf16* __restrict__ fb,
                                                     const float* __restrict__ bias,
                                                     float* __restrict__ y) {
  __shared__ __bf16 sA[2][64][32];    // 8 KB
  __shared__ __bf16 sB[2][128][32];   // 16 KB

  const int t    = threadIdx.x;
  const int lane = t & 31;
  const int wv   = t >> 5;
  const int lm   = lane >> 4, ln = lane & 15;
  const int wm   = wv >> 2;          // 0..1  (M)
  const int wn   = wv & 3;           // 0..3  (N)
  const int rowBase = blockIdx.x * 64;
  const int colBase = blockIdx.y * 128;

  const __bf16* gA = xb + (size_t)rowBase * DD;   // 64 rows
  const __bf16* gB = fb + (size_t)colBase * DD;   // 128 rows

#if USE_TDM
  if (wv == 0) {
    tdm_load_tile(gA, (unsigned)(size_t)&sA[0][0][0], 64);
    tdm_load_tile(gB, (unsigned)(size_t)&sB[0][0][0], 128);
    __builtin_amdgcn_s_wait_tensorcnt(0);
  }
#else
  {
    const int r = t >> 2, c = (t & 3) * 8;
    *(v8bf*)&sA[0][r][c] = *(const v8bf*)(gA + (size_t)r * DD + c);
    const int r2 = (t + 256) >> 2, c2 = ((t + 256) & 3) * 8;
    *(v8bf*)&sB[0][r][c]  = *(const v8bf*)(gB + (size_t)r * DD + c);
    *(v8bf*)&sB[0][r2][c2] = *(const v8bf*)(gB + (size_t)r2 * DD + c2);
  }
#endif
  __syncthreads();

  v8f acc00 = {}, acc01 = {}, acc10 = {}, acc11 = {};
  int s = 0;
#pragma unroll 1
  for (int k = 0; k < DD; k += 32, s ^= 1) {
    const int kn = k + 32;
    if (kn < DD) {
#if USE_TDM
      if (wv == 0) {
        tdm_load_tile(gA + kn, (unsigned)(size_t)&sA[s ^ 1][0][0], 64);
        tdm_load_tile(gB + kn, (unsigned)(size_t)&sB[s ^ 1][0][0], 128);
      }
#else
      const int r = t >> 2, c = (t & 3) * 8;
      *(v8bf*)&sA[s ^ 1][r][c] = *(const v8bf*)(gA + (size_t)r * DD + kn + c);
      const int r2 = (t + 256) >> 2, c2 = ((t + 256) & 3) * 8;
      *(v8bf*)&sB[s ^ 1][r][c]  = *(const v8bf*)(gB + (size_t)r * DD + kn + c);
      *(v8bf*)&sB[s ^ 1][r2][c2] = *(const v8bf*)(gB + (size_t)r2 * DD + kn + c2);
#endif
    }
    // compute on stage s from LDS (ds_load_b128 fragments)
    v16bf a0 = load_a_bf16(&sA[s][wm * 32 + ln][0], lm);
    v16bf a1 = load_a_bf16(&sA[s][wm * 32 + 16 + ln][0], lm);
    v16bf b0 = load_b_bf16(&sB[s][wn * 32 + ln][0], lm);
    v16bf b1 = load_b_bf16(&sB[s][wn * 32 + 16 + ln][0], lm);
    acc00 = __builtin_amdgcn_wmma_f32_16x16x32_bf16(false, a0, false, b0, (short)0, acc00, false, false);
    acc01 = __builtin_amdgcn_wmma_f32_16x16x32_bf16(false, a0, false, b1, (short)0, acc01, false, false);
    acc10 = __builtin_amdgcn_wmma_f32_16x16x32_bf16(false, a1, false, b0, (short)0, acc10, false, false);
    acc11 = __builtin_amdgcn_wmma_f32_16x16x32_bf16(false, a1, false, b1, (short)0, acc11, false, false);
#if USE_TDM
    if (wv == 0) __builtin_amdgcn_s_wait_tensorcnt(0);
#endif
    __syncthreads();
  }

  v8f acc[2][2] = { { acc00, acc01 }, { acc10, acc11 } };
#pragma unroll
  for (int j = 0; j < 2; ++j) {
    const float bj = bias[colBase + wn * 32 + j * 16 + ln];
#pragma unroll
    for (int i = 0; i < 2; ++i) {
      float* out = y + (size_t)(rowBase + wm * 32 + i * 16 + lm * 8) * OO
                     + (colBase + wn * 32 + j * 16 + ln);
#pragma unroll
      for (int r = 0; r < 8; ++r) out[(size_t)r * OO] = acc[i][j][r] + bj;
    }
  }
}

// ---------------- sd = gather(x, idx) @ Qᵀ  (f32 WMMA, exact) ----------------
__global__ void sd_gemm_kernel(const float* __restrict__ x,
                               const float* __restrict__ Q,
                               const int* __restrict__ idx,
                               float* __restrict__ sd) {
  const int lane = threadIdx.x & 31;
  const int wv   = threadIdx.x >> 5;
  const int lm   = lane >> 4, ln = lane & 15;
  const int row0 = blockIdx.x * 64  + (wv >> 2) * 32;
  const int col0 = blockIdx.y * 128 + (wv & 3) * 32;

  const int g0 = row0 + ln;
  const int g1 = g0 + 16;
  const float* xr0 = x + ((size_t)(g0 / NSAMP) * LL + (size_t)idx[g0]) * DD;
  const float* xr1 = x + ((size_t)(g1 / NSAMP) * LL + (size_t)idx[g1]) * DD;
  const float* qr0 = Q + (size_t)(col0 + ln) * DD;
  const float* qr1 = qr0 + (size_t)16 * DD;

  v8f acc00 = {}, acc01 = {}, acc10 = {}, acc11 = {};
#pragma unroll 2
  for (int k = 0; k < DD; k += 4) {
    const int ka = k + lm * 2;
    v2f a0 = *(const v2f*)(xr0 + ka);
    v2f a1 = *(const v2f*)(xr1 + ka);
    v2f b0 = *(const v2f*)(qr0 + ka);
    v2f b1 = *(const v2f*)(qr1 + ka);
    acc00 = __builtin_amdgcn_wmma_f32_16x16x4_f32(false, a0, false, b0, (short)0, acc00, false, false);
    acc01 = __builtin_amdgcn_wmma_f32_16x16x4_f32(false, a0, false, b1, (short)0, acc01, false, false);
    acc10 = __builtin_amdgcn_wmma_f32_16x16x4_f32(false, a1, false, b0, (short)0, acc10, false, false);
    acc11 = __builtin_amdgcn_wmma_f32_16x16x4_f32(false, a1, false, b1, (short)0, acc11, false, false);
  }

  v8f acc[2][2] = { { acc00, acc01 }, { acc10, acc11 } };
#pragma unroll
  for (int i = 0; i < 2; ++i)
#pragma unroll
    for (int j = 0; j < 2; ++j) {
      float* out = sd + (size_t)(row0 + i * 16 + lm * 8) * DD + (col0 + j * 16 + ln);
#pragma unroll
      for (int r = 0; r < 8; ++r) out[(size_t)r * DD] = acc[i][j][r];
    }
}

// ---------------- grad from M = sdᵀ sd / N  (f32 WMMA) ----------------
__global__ void m_gemm_kernel(const float* __restrict__ sd, float* __restrict__ grad) {
  const int lane = threadIdx.x & 31;
  const int wv   = threadIdx.x >> 5;
  const int lm   = lane >> 4, ln = lane & 15;
  const int row0 = blockIdx.x * 64  + (wv >> 2) * 32;
  const int col0 = blockIdx.y * 128 + (wv & 3) * 32;

  v8f acc00 = {}, acc01 = {}, acc10 = {}, acc11 = {};
#pragma unroll 1
  for (int k = 0; k < NR; k += 4) {
    const int ka = k + lm * 2;
    const float* base = sd + (size_t)ka * DD;
    v2f a0 = { base[row0 + ln],      base[DD + row0 + ln] };
    v2f a1 = { base[row0 + 16 + ln], base[DD + row0 + 16 + ln] };
    v2f b0 = { base[col0 + ln],      base[DD + col0 + ln] };
    v2f b1 = { base[col0 + 16 + ln], base[DD + col0 + 16 + ln] };
    acc00 = __builtin_amdgcn_wmma_f32_16x16x4_f32(false, a0, false, b0, (short)0, acc00, false, false);
    acc01 = __builtin_amdgcn_wmma_f32_16x16x4_f32(false, a0, false, b1, (short)0, acc01, false, false);
    acc10 = __builtin_amdgcn_wmma_f32_16x16x4_f32(false, a1, false, b0, (short)0, acc10, false, false);
    acc11 = __builtin_amdgcn_wmma_f32_16x16x4_f32(false, a1, false, b1, (short)0, acc11, false, false);
  }

  const float invN = 1.0f / (float)NR;
  v8f acc[2][2] = { { acc00, acc01 }, { acc10, acc11 } };
#pragma unroll
  for (int i = 0; i < 2; ++i)
#pragma unroll
    for (int j = 0; j < 2; ++j)
#pragma unroll
      for (int r = 0; r < 8; ++r) {
        const int gi = row0 + i * 16 + lm * 8 + r;
        const int gj = col0 + j * 16 + ln;
        const float v = acc[i][j][r] * invN;
        const float g = (gi == gj) ? (KAPPA * (v - 1.0f)) : ((1.0f - KAPPA) * v);
        grad[(size_t)gi * DD + gj] = g;
      }
}

// ---------------- per-row loss partials (deterministic tree) ----------------
__global__ void row_stats_kernel(const float* __restrict__ sd,
                                 float* __restrict__ rowA, float* __restrict__ rowW) {
  const int t = threadIdx.x;
  const float* p = sd + (size_t)blockIdx.x * DD;
  float n2 = 0.f, s4 = 0.f, w = 0.f;
  for (int c = t; c < DD; c += 256) {
    const float v = p[c];
    const float s2 = v * v;
    n2 += s2;
    s4 += s2 * s2;
    const float d = s2 - 1.0f;
    w += d * d;
  }
  __shared__ float sA[256], sB[256], sC[256];
  sA[t] = n2; sB[t] = s4; sC[t] = w;
  __syncthreads();
  for (int s = 128; s > 0; s >>= 1) {
    if (t < s) { sA[t] += sA[t + s]; sB[t] += sB[t + s]; sC[t] += sC[t + s]; }
    __syncthreads();
  }
  if (t == 0) {
    rowA[blockIdx.x] = sA[0] * sA[0] - sB[0];
    rowW[blockIdx.x] = sC[0];
  }
}

__global__ void finalize_loss_kernel(const float* __restrict__ rowA,
                                     const float* __restrict__ rowW,
                                     float* __restrict__ out) {
  const int t = threadIdx.x;
  float a = 0.f, b = 0.f;
  for (int i = t; i < NR; i += 256) { a += rowA[i]; b += rowW[i]; }
  __shared__ float sA[256], sB[256];
  sA[t] = a; sB[t] = b;
  __syncthreads();
  for (int s = 128; s > 0; s >>= 1) {
    if (t < s) { sA[t] += sA[t + s]; sB[t] += sB[t + s]; }
    __syncthreads();
  }
  if (t == 0) {
    const float inv_dd = 1.0f / ((float)DD * (float)DD);
    out[0] = (sA[0] / (float)NR) * inv_dd;
    out[1] = (sB[0] / (float)NR) * inv_dd;
  }
}

extern "C" void kernel_launch(void* const* d_in, const int* in_sizes, int n_in,
                              void* d_out, int out_size, void* d_ws, size_t ws_size,
                              hipStream_t stream) {
  (void)in_sizes; (void)n_in; (void)out_size; (void)ws_size;
  const float* x    = (const float*)d_in[0];
  const float* w    = (const float*)d_in[1];
  const float* bias = (const float*)d_in[2];
  const float* q    = (const float*)d_in[3];
  const int*   idx  = (const int*)d_in[4];

  float* y      = (float*)d_out;
  float* grad   = y + (size_t)MROWS * OO;
  float* losses = grad + (size_t)DD * DD;

  char* ws = (char*)d_ws;
  __bf16* fusedB = (__bf16*)ws;                                                 // 4 MB
  __bf16* xB     = (__bf16*)(ws + (size_t)OO * DD * 2);                         // 64 MB
  float*  sd     = (float*)(ws + (size_t)OO * DD * 2 + (size_t)MROWS * DD * 2); // 32 MB
  float*  rowA   = sd + (size_t)NR * DD;
  float*  rowW   = rowA + NR;

  convert_x_kernel<<<(MROWS * (size_t)DD) / (256 * 8), 256, 0, stream>>>(x, xB);
  fused_gemm_kernel<<<dim3(OO / 128, DD / 64), 256, 0, stream>>>(w, q, fusedB);
  y_gemm_kernel<<<dim3(MROWS / 64, OO / 128), 256, 0, stream>>>(xB, fusedB, bias, y);
  sd_gemm_kernel<<<dim3(NR / 64, DD / 128), 256, 0, stream>>>(x, q, idx, sd);
  m_gemm_kernel<<<dim3(DD / 64, DD / 128), 256, 0, stream>>>(sd, grad);
  row_stats_kernel<<<NR, 256, 0, stream>>>(sd, rowA, rowW);
  finalize_loss_kernel<<<1, 256, 0, stream>>>(rowA, rowW, losses);
}
